// GLPDepth_add_meanhr_79774722556570
// MI455X (gfx1250) — matrix operand order repaired and checked by reading, not compiled, not run on hardware
//
#include <hip/hip_runtime.h>
#include <hip/hip_bf16.h>
#include <math.h>

// ---------------------------------------------------------------------------
// GLPDepth decoder for MI455X (gfx1250, wave32, WMMA + TDM).
// Conv layers = implicit GEMM on v_wmma_f32_16x16x32_bf16:
//   A (16 out-ch x 32 K) : weights, whole 16 x Ktot tile DMA'd to LDS once per
//                          block via tensor_load_to_lds (TDM), zero inner barriers
//   B (32 K x 16 pixels) : bf16 activations gathered from global (2 tiles/wave,
//                          A-fragment reused across both WMMAs)
//   C/D (16x16 f32)      : accumulators, fused bias/BN/ReLU/sigmoid epilogue
// ---------------------------------------------------------------------------

typedef __bf16 bf16_t;
typedef __attribute__((ext_vector_type(16))) __bf16 v16bf;
typedef __attribute__((ext_vector_type(8)))  float  v8f;
typedef __attribute__((ext_vector_type(4)))  unsigned int v4u;
typedef __attribute__((ext_vector_type(4)))  int v4i;
typedef __attribute__((ext_vector_type(8)))  int v8i;

#define BATCH   2
#define FFRAMES 8
#define MASK_W  384
#define MASK_HW (384 * 384)

#if defined(__has_builtin)
# if __has_builtin(__builtin_amdgcn_tensor_load_to_lds)
#  define HAVE_TDM 1
# endif
#endif
#ifndef HAVE_TDM
# define HAVE_TDM 0
#endif

__device__ inline bf16_t f2b(float f) {
  union { float f; unsigned u; } v; v.f = f;
  unsigned r = v.u + 0x7FFFu + ((v.u >> 16) & 1u);      // round-nearest-even
  union { unsigned short s; bf16_t b; } o;
  o.s = (unsigned short)(r >> 16);
  return o.b;
}

__device__ inline void st_out(float* d, size_t i, float v)  { d[i] = v; }
__device__ inline void st_out(bf16_t* d, size_t i, float v) { d[i] = f2b(v); }

// ---------------------------------------------------------------------------
// WMMA implicit-GEMM convolution (1x1 or 3x3, stride 1, pad = taps==9 ? 1 : 0)
//   X : bf16 [BATCH][Cin][H*W]
//   Wt: bf16 [CoutPad][taps][Cin]   (zero padded rows for co >= Cout)
//   Y : f32  [BATCH][CoutStore][H*W]
// mode: 0 = +bias   1 = +bias,BN,ReLU   2 = +bias,sigmoid   3 = +bias,10*sigmoid
// Dynamic LDS: 16 * Ktot bf16 (weight tile for this block's 16 out-channels).
// ---------------------------------------------------------------------------
__global__ __launch_bounds__(128) void k_conv_wmma(
    const bf16_t* __restrict__ X, int Cin, int H, int W,
    const bf16_t* __restrict__ Wt, int taps,
    const float* __restrict__ bias, int Cout,
    float* __restrict__ Y, int CoutStore, int mode,
    const float* __restrict__ scale, const float* __restrict__ shift)
{
  extern __shared__ bf16_t sW[];        // [16][Ktot] weight tile
  const int P      = H * W;
  const int lane   = threadIdx.x & 31;
  const int wave   = threadIdx.x >> 5;
  const int half   = lane >> 4;
  const int l16    = lane & 15;
  const int coBase = blockIdx.y * 16;
  const int b      = blockIdx.z;
  const int Ktot   = taps * Cin;
  const bf16_t* Xb    = X + (size_t)b * Cin * P;
  const bf16_t* Wtile = Wt + (size_t)coBase * Ktot;

#if HAVE_TDM
  // --- TDM: DMA the full 16 x Ktot bf16 weight tile into LDS (wave 0 only) ---
  if (wave == 0) {
    const unsigned lds_off = (unsigned)(size_t)(void*)sW;       // LDS aperture low bits
    const unsigned long long ga = (unsigned long long)(size_t)Wtile;
    v4u g0;
    g0[0] = 1u;                                                 // count=1, user mode
    g0[1] = lds_off;                                            // lds_addr
    g0[2] = (unsigned)(ga & 0xFFFFFFFFu);                       // global_addr[31:0]
    g0[3] = (unsigned)((ga >> 32) & 0x01FFFFFFu) | (2u << 30);  // addr[56:32] | type=2
    v8i g1;
    g1[0] = (int)(1u << 16);                       // wg_mask=0, data_size=2B
    g1[1] = (int)(((unsigned)Ktot & 0xFFFFu) << 16);  // tensor_dim0[15:0]
    g1[2] = (int)(((unsigned)Ktot >> 16) | (16u << 16)); // dim0[31:16] | tensor_dim1=16
    g1[3] = (int)(((unsigned)Ktot & 0xFFFFu) << 16);  // tile_dim0 = Ktot
    g1[4] = 16;                                    // tile_dim1 = 16, tile_dim2 = 0
    g1[5] = Ktot;                                  // tensor_dim0_stride[31:0]
    g1[6] = 0;                                     // stride[47:32] | dim1_stride lo
    g1[7] = 0;
    v4i z4 = {0, 0, 0, 0};
#if __clang_major__ >= 23
    v8i z8 = {0, 0, 0, 0, 0, 0, 0, 0};
    __builtin_amdgcn_tensor_load_to_lds(g0, g1, z4, z4, z8, 0);
#else
    __builtin_amdgcn_tensor_load_to_lds(g0, g1, z4, z4, 0);
#endif
    __builtin_amdgcn_s_wait_tensorcnt(0);
  }
#else
  for (int t = threadIdx.x; t < 16 * Ktot; t += 128)
    sW[t] = Wtile[t];
#endif
  __syncthreads();

  // two 16-pixel tiles per wave; A fragment reused across both WMMAs
  const int nTile0 = blockIdx.x * 8 + wave * 2;
  const int pB[2] = { nTile0 * 16, nTile0 * 16 + 16 };
  v8f acc[2] = { v8f{}, v8f{} };
  int  py_[2], px_[2];
  bool valid[2];
#pragma unroll
  for (int t = 0; t < 2; ++t) {
    const int mp = pB[t] + l16;
    valid[t] = mp < P;
    const int pcl = valid[t] ? mp : (P - 1);
    py_[t] = pcl / W;
    px_[t] = pcl % W;
  }

  for (int tap = 0; tap < taps; ++tap) {
    const int dy = (taps == 9) ? (tap / 3 - 1) : 0;
    const int dx = (taps == 9) ? (tap % 3 - 1) : 0;
    bool inb[2]; size_t sb[2];
#pragma unroll
    for (int t = 0; t < 2; ++t) {
      const int yy = py_[t] + dy, xx = px_[t] + dx;
      inb[t] = valid[t] & (yy >= 0) & (yy < H) & (xx >= 0) & (xx < W);
      sb[t] = inb[t] ? (size_t)(yy * W + xx) : 0;
    }

    for (int kc = 0; kc < Cin; kc += 32) {
      const int kBase = tap * Cin + kc;
      v16bf afrag;
#pragma unroll
      for (int j = 0; j < 16; ++j) {
        // A 16x32 bf16 layout: lanes 0-15 M=0..15; v0-3 K=half*8+j, v4-7 K=16+half*8+(j-8)
        const int ka = (j < 8) ? (half * 8 + j) : (16 + half * 8 + (j - 8));
        afrag[j] = sW[l16 * Ktot + kBase + ka];
      }
#pragma unroll
      for (int t = 0; t < 2; ++t) {
        const bf16_t* xcol = Xb + (size_t)kc * P + sb[t];
        __builtin_prefetch(xcol + 32 * (size_t)P, 0, 1);   // next K chunk
        v16bf bfrag;
#pragma unroll
        for (int j = 0; j < 16; ++j) {
          // B 32x16 bf16 layout: N = lane&15, K = half*8+j (j<8), 16+half*8+(j-8) (j>=8)
          const int kb = (j < 8) ? (half * 8 + j) : (16 + half * 8 + (j - 8));
          bf16_t v = xcol[(size_t)kb * P];
          bfrag[j] = inb[t] ? v : f2b(0.0f);
        }
        acc[t] = __builtin_amdgcn_wmma_f32_16x16x32_bf16(
            false, afrag, false, bfrag, (short)0, acc[t], false, false);
      }
    }
  }

  // C/D 16x16 f32 layout: VGPR r -> M = r + 8*(lane>>4), N = lane&15
#pragma unroll
  for (int t = 0; t < 2; ++t) {
    const int pOut = pB[t] + l16;
    if (pOut < P) {
#pragma unroll
      for (int r = 0; r < 8; ++r) {
        const int co = coBase + r + 8 * half;
        if (co < CoutStore) {
          float y = acc[t][r] + ((co < Cout) ? bias[co] : 0.0f);
          if (mode == 1)      { y = y * scale[co] + shift[co]; y = fmaxf(y, 0.0f); }
          else if (mode == 2) { y = 1.0f  / (1.0f + __expf(-y)); }
          else if (mode == 3) { y = 10.0f / (1.0f + __expf(-y)); }
          Y[((size_t)b * CoutStore + co) * P + pOut] = y;
        }
      }
    }
  }
}

// ---------------------------------------------------------------------------
// OIHW f32 weights -> [CoutPad][tap][ci] bf16 (zero pad co >= Cout)
// ---------------------------------------------------------------------------
__global__ void k_wcvt(const float* __restrict__ w, bf16_t* __restrict__ dst,
                       int Cout, int CoutPad, int Cin, int taps)
{
  const int Ktot = Cin * taps;
  const int n = CoutPad * Ktot;
  const int i = blockIdx.x * blockDim.x + threadIdx.x;
  if (i >= n) return;
  const int k = i % Ktot, co = i / Ktot;
  const int ci = k % Cin, tap = k / Cin;
  const float v = (co < Cout) ? w[((size_t)co * Cin + ci) * taps + tap] : 0.0f;
  dst[i] = f2b(v);
}

__global__ void k_bn_prep(const float* __restrict__ g, const float* __restrict__ bb,
                          const float* __restrict__ rm, const float* __restrict__ rv,
                          float* __restrict__ scale, float* __restrict__ shift, int C)
{
  const int i = blockIdx.x * blockDim.x + threadIdx.x;
  if (i >= C) return;
  const float s = g[i] * rsqrtf(rv[i] + 1e-5f);
  scale[i] = s;
  shift[i] = bb[i] - rm[i] * s;
}

__global__ void k_cvt(const float* __restrict__ s, bf16_t* __restrict__ d, int n)
{
  const int i = blockIdx.x * blockDim.x + threadIdx.x;
  if (i < n) d[i] = f2b(s[i]);
}

__global__ void k_relu_cvt(const float* __restrict__ s, bf16_t* __restrict__ d, int n)
{
  const int i = blockIdx.x * blockDim.x + threadIdx.x;
  if (i < n) d[i] = f2b(fmaxf(s[i], 0.0f));
}

// concat([local C1, global C2], ch) -> bf16
__global__ void k_cat_cvt(const float* __restrict__ a, const float* __restrict__ g,
                          bf16_t* __restrict__ dst, int C1, int C2, int P)
{
  const int Ct = C1 + C2;
  const size_t n = (size_t)BATCH * Ct * P;
  const size_t i = (size_t)blockIdx.x * blockDim.x + threadIdx.x;
  if (i >= n) return;
  const int p = i % P;
  const int c = (int)((i / P) % Ct);
  const int b = (int)(i / ((size_t)Ct * P));
  const float v = (c < C1) ? a[((size_t)b * C1 + c) * P + p]
                           : g[((size_t)b * C2 + (c - C1)) * P + p];
  dst[i] = f2b(v);
}

// out = x_local * attn[ch0] + x_global * attn[ch1]   (attn already sigmoided)
__global__ void k_sff_comb(const float* __restrict__ xl, const float* __restrict__ xg,
                           const float* __restrict__ attn, float* __restrict__ out, int P)
{
  const size_t n = (size_t)BATCH * 64 * P;
  const size_t i = (size_t)blockIdx.x * blockDim.x + threadIdx.x;
  if (i >= n) return;
  const int p = i % P;
  const int b = (int)(i / ((size_t)64 * P));
  const float a0 = attn[((size_t)b * 16 + 0) * P + p];
  const float a1 = attn[((size_t)b * 16 + 1) * P + p];
  out[i] = xl[i] * a0 + xg[i] * a1;
}

// jax.image.resize bilinear x2 (half-pixel centers, clamp-to-edge)
template <typename OT>
__global__ void k_up2(const float* __restrict__ src, OT* __restrict__ dst,
                      int C, int H, int W)
{
  const int oH = H * 2, oW = W * 2;
  const size_t n = (size_t)BATCH * C * oH * oW;
  const size_t i = (size_t)blockIdx.x * blockDim.x + threadIdx.x;
  if (i >= n) return;
  const int ox = i % oW;
  const int oy = (int)((i / oW) % oH);
  const int c  = (int)((i / ((size_t)oW * oH)) % C);
  const int b  = (int)(i / ((size_t)oW * oH * C));
  const float fy = (oy + 0.5f) * 0.5f - 0.5f;
  const float fx = (ox + 0.5f) * 0.5f - 0.5f;
  const int y0 = (int)floorf(fy), x0 = (int)floorf(fx);
  const float wy = fy - y0, wx = fx - x0;
  const int y0c = y0 < 0 ? 0 : (y0 >= H ? H - 1 : y0);
  const int y1c = (y0 + 1) >= H ? H - 1 : (y0 + 1 < 0 ? 0 : y0 + 1);
  const int x0c = x0 < 0 ? 0 : (x0 >= W ? W - 1 : x0);
  const int x1c = (x0 + 1) >= W ? W - 1 : (x0 + 1 < 0 ? 0 : x0 + 1);
  const float* S = src + ((size_t)b * C + c) * H * W;
  const float v00 = S[y0c * W + x0c], v01 = S[y0c * W + x1c];
  const float v10 = S[y1c * W + x0c], v11 = S[y1c * W + x1c];
  const float v = (1.0f - wy) * ((1.0f - wx) * v00 + wx * v01) +
                  wy          * ((1.0f - wx) * v10 + wx * v11);
  st_out(dst, i, v);
}

// masked blend, phase 1: sums[b*C+c] = sum_p mask[b,frame,y*s,x*s] * out[b,c,p]
__global__ __launch_bounds__(256) void k_blend_reduce(
    const float* __restrict__ out, const float* __restrict__ mask, int frame,
    float* __restrict__ sums, int C, int H, int W, int s)
{
  const int bc = blockIdx.x;
  const int b = bc / C, c = bc % C;
  const float* M = mask + ((size_t)b * FFRAMES + frame) * MASK_HW;
  const float* O = out + ((size_t)b * C + c) * H * W;
  float acc = 0.0f;
  for (int p = threadIdx.x; p < H * W; p += 256) {
    const int y = p / W, x = p % W;
    acc += M[(y * s) * MASK_W + x * s] * O[p];
  }
  __shared__ float red[256];
  red[threadIdx.x] = acc;
  __syncthreads();
  for (int st = 128; st > 0; st >>= 1) {
    if (threadIdx.x < st) red[threadIdx.x] += red[threadIdx.x + st];
    __syncthreads();
  }
  if (threadIdx.x == 0) sums[bc] = red[0];
}

// masked blend, phase 2: out = out*(1-m) + (sum/mx)*m
__global__ void k_blend_apply(float* __restrict__ out, const float* __restrict__ mask,
                              int frame, const float* __restrict__ sums,
                              const float* __restrict__ mxl, int lvl,
                              int C, int H, int W, int s)
{
  const size_t n = (size_t)BATCH * C * H * W;
  const size_t i = (size_t)blockIdx.x * blockDim.x + threadIdx.x;
  if (i >= n) return;
  const int p = i % (H * W);
  const int c = (int)((i / (H * W)) % C);
  const int b = (int)(i / ((size_t)C * H * W));
  const int y = p / W, x = p % W;
  const float m = mask[((size_t)b * FFRAMES + frame) * MASK_HW + (y * s) * MASK_W + x * s];
  const float mx = mxl[b * 40 + lvl * 8 + frame];     // mx_len (B,1,5,8)
  const float mean = sums[b * C + c] / mx;
  const float v = out[i];
  out[i] = v * (1.0f - m) + mean * m;
}

// ---------------------------------------------------------------------------
extern "C" void kernel_launch(void* const* d_in, const int* in_sizes, int n_in,
                              void* d_out, int out_size, void* d_ws, size_t ws_size,
                              hipStream_t stream)
{
  (void)in_sizes; (void)n_in; (void)out_size; (void)ws_size;
  const int P12 = 144, P24 = 576, P48 = 2304, P96 = 9216, P192 = 36864, P384 = 147456;

  const float* x1   = (const float*)d_in[0];
  const float* x2   = (const float*)d_in[1];
  const float* x3   = (const float*)d_in[2];
  const float* x4   = (const float*)d_in[3];
  const float* mask = (const float*)d_in[4];
  const float* mxl  = (const float*)d_in[5];
#define PF(i) ((const float*)d_in[(i)])
  // params leaves (sorted-key pytree order):
  // 6:bot.b 7:bot.w  8..21:fus1  22..35:fus2  36..49:fus3
  //   fus: +0 b1 +1 b2 +2 b3 +3 bn1.b +4 bn1.g +5 bn1.rm +6 bn1.rv
  //        +7 bn2.b +8 bn2.g +9 bn2.rm +10 bn2.rv +11 w1 +12 w2 +13 w3
  // 50:last1.b 51:last1.w 52:last2.b 53:last2.w 54:sk1.b 55:sk1.w 56:sk2.b 57:sk2.w

  // ---- workspace bump allocator ----
  char* wsp = (char*)d_ws;
  size_t off = 0;
  auto allocB = [&](size_t e) { bf16_t* p = (bf16_t*)(wsp + off);
                                off = (off + e * 2 + 255) & ~(size_t)255; return p; };
  auto allocF = [&](size_t e) { float* p = (float*)(wsp + off);
                                off = (off + e * 4 + 255) & ~(size_t)255; return p; };

  bf16_t* wbBot = allocB(64 * 512);
  bf16_t* wbSk1 = allocB(64 * 320);
  bf16_t* wbSk2 = allocB(64 * 128);
  bf16_t* wbFus[3][3];
  float *sc1[3], *sh1[3], *sc2[3], *sh2[3];
  for (int f = 0; f < 3; ++f) {
    wbFus[f][0] = allocB(64 * 1152);
    wbFus[f][1] = allocB(32 * 576);
    wbFus[f][2] = allocB(16 * 288);
    sc1[f] = allocF(64); sh1[f] = allocF(64);
    sc2[f] = allocF(32); sh2[f] = allocF(32);
  }
  bf16_t* wbL1 = allocB(64 * 576);
  bf16_t* wbL2 = allocB(16 * 576);

  bf16_t* x4b = allocB((size_t)BATCH * 512 * P12);
  bf16_t* x3b = allocB((size_t)BATCH * 320 * P24);
  bf16_t* x2b = allocB((size_t)BATCH * 128 * P48);

  float*  tBot  = allocF((size_t)BATCH * 64 * P12);
  float*  g24   = allocF((size_t)BATCH * 64 * P24);
  float*  x3s   = allocF((size_t)BATCH * 64 * P24);
  bf16_t* cat24 = allocB((size_t)BATCH * 128 * P24);
  float*  sA24  = allocF((size_t)BATCH * 64 * P24);
  bf16_t* sA24b = allocB((size_t)BATCH * 64 * P24);
  float*  sB24  = allocF((size_t)BATCH * 32 * P24);
  bf16_t* sB24b = allocB((size_t)BATCH * 32 * P24);
  float*  at24  = allocF((size_t)BATCH * 16 * P24);
  float*  fu24  = allocF((size_t)BATCH * 64 * P24);

  float*  g48   = allocF((size_t)BATCH * 64 * P48);
  float*  x2s   = allocF((size_t)BATCH * 64 * P48);
  bf16_t* cat48 = allocB((size_t)BATCH * 128 * P48);
  float*  sA48  = allocF((size_t)BATCH * 64 * P48);
  bf16_t* sA48b = allocB((size_t)BATCH * 64 * P48);
  float*  sB48  = allocF((size_t)BATCH * 32 * P48);
  bf16_t* sB48b = allocB((size_t)BATCH * 32 * P48);
  float*  at48  = allocF((size_t)BATCH * 16 * P48);
  float*  fu48  = allocF((size_t)BATCH * 64 * P48);

  float*  g96   = allocF((size_t)BATCH * 64 * P96);
  bf16_t* cat96 = allocB((size_t)BATCH * 128 * P96);
  float*  sA96  = allocF((size_t)BATCH * 64 * P96);
  bf16_t* sA96b = allocB((size_t)BATCH * 64 * P96);
  float*  sB96  = allocF((size_t)BATCH * 32 * P96);
  bf16_t* sB96b = allocB((size_t)BATCH * 32 * P96);
  float*  at96  = allocF((size_t)BATCH * 16 * P96);
  float*  fu96  = allocF((size_t)BATCH * 64 * P96);

  float*  out192   = allocF((size_t)BATCH * 64 * P192);
  bf16_t* out384b  = allocB((size_t)BATCH * 64 * P384);
  float*  conv384  = allocF((size_t)BATCH * 64 * P384);
  bf16_t* relu384b = allocB((size_t)BATCH * 64 * P384);
  float*  bsum     = allocF(BATCH * 64);

  auto eg = [](size_t n) { return dim3((unsigned)((n + 255) / 256)); };
  auto conv = [&](const bf16_t* X, int Cin, int H, int W, const bf16_t* Wt,
                  int CoutPad, int taps, const float* bias, int Cout,
                  float* Y, int CoutStore, int mode,
                  const float* sc, const float* sh) {
    const int P = H * W;
    const int Ktot = taps * Cin;
    dim3 g((unsigned)((((P + 15) / 16) + 7) / 8), (unsigned)(CoutPad / 16), BATCH);
    k_conv_wmma<<<g, 128, (size_t)16 * Ktot * sizeof(bf16_t), stream>>>(
        X, Cin, H, W, Wt, taps, bias, Cout, Y, CoutStore, mode, sc, sh);
  };

  // ---- weight conversion + BN folding ----
  k_wcvt<<<eg(64 * 512), 256, 0, stream>>>(PF(7),  wbBot, 64, 64, 512, 1);
  k_wcvt<<<eg(64 * 320), 256, 0, stream>>>(PF(55), wbSk1, 64, 64, 320, 1);
  k_wcvt<<<eg(64 * 128), 256, 0, stream>>>(PF(57), wbSk2, 64, 64, 128, 1);
  for (int f = 0; f < 3; ++f) {
    const int fb = 8 + 14 * f;
    k_wcvt<<<eg(64 * 1152), 256, 0, stream>>>(PF(fb + 11), wbFus[f][0], 64, 64, 128, 9);
    k_wcvt<<<eg(32 * 576),  256, 0, stream>>>(PF(fb + 12), wbFus[f][1], 32, 32, 64, 9);
    k_wcvt<<<eg(16 * 288),  256, 0, stream>>>(PF(fb + 13), wbFus[f][2], 2, 16, 32, 9);
    k_bn_prep<<<1, 64, 0, stream>>>(PF(fb + 4), PF(fb + 3), PF(fb + 5), PF(fb + 6),
                                    sc1[f], sh1[f], 64);
    k_bn_prep<<<1, 32, 0, stream>>>(PF(fb + 8), PF(fb + 7), PF(fb + 9), PF(fb + 10),
                                    sc2[f], sh2[f], 32);
  }
  k_wcvt<<<eg(64 * 576), 256, 0, stream>>>(PF(51), wbL1, 64, 64, 64, 9);
  k_wcvt<<<eg(16 * 576), 256, 0, stream>>>(PF(53), wbL2, 1, 16, 64, 9);

  k_cvt<<<eg((size_t)BATCH * 512 * P12), 256, 0, stream>>>(x4, x4b, BATCH * 512 * P12);
  k_cvt<<<eg((size_t)BATCH * 320 * P24), 256, 0, stream>>>(x3, x3b, BATCH * 320 * P24);
  k_cvt<<<eg((size_t)BATCH * 128 * P48), 256, 0, stream>>>(x2, x2b, BATCH * 128 * P48);

  // ---- SFF block ----
  auto run_sff = [&](const float* xl, const float* xg, int H, int W, int f,
                     bf16_t* catb, float* sa, bf16_t* sab, float* sb, bf16_t* sbb,
                     float* attn, float* fout) {
    const int P = H * W;
    k_cat_cvt<<<eg((size_t)BATCH * 128 * P), 256, 0, stream>>>(xl, xg, catb, 64, 64, P);
    conv(catb, 128, H, W, wbFus[f][0], 64, 9, PF(8 + 14 * f + 0), 64,
         sa, 64, 1, sc1[f], sh1[f]);
    k_cvt<<<eg((size_t)BATCH * 64 * P), 256, 0, stream>>>(sa, sab, BATCH * 64 * P);
    conv(sab, 64, H, W, wbFus[f][1], 32, 9, PF(8 + 14 * f + 1), 32,
         sb, 32, 1, sc2[f], sh2[f]);
    k_cvt<<<eg((size_t)BATCH * 32 * P), 256, 0, stream>>>(sb, sbb, BATCH * 32 * P);
    conv(sbb, 32, H, W, wbFus[f][2], 16, 9, PF(8 + 14 * f + 2), 2,
         attn, 16, 2, nullptr, nullptr);
    k_sff_comb<<<eg((size_t)BATCH * 64 * P), 256, 0, stream>>>(xl, xg, attn, fout, P);
  };

  // ---- decoder pipeline ----
  conv(x4b, 512, 12, 12, wbBot, 64, 1, PF(6), 64, tBot, 64, 0, nullptr, nullptr);
  k_up2<float><<<eg((size_t)BATCH * 64 * P24), 256, 0, stream>>>(tBot, g24, 64, 12, 12);

  conv(x3b, 320, 24, 24, wbSk1, 64, 1, PF(54), 64, x3s, 64, 0, nullptr, nullptr);
  run_sff(x3s, g24, 24, 24, 0, cat24, sA24, sA24b, sB24, sB24b, at24, fu24);
  k_up2<float><<<eg((size_t)BATCH * 64 * P48), 256, 0, stream>>>(fu24, g48, 64, 24, 24);

  conv(x2b, 128, 48, 48, wbSk2, 64, 1, PF(56), 64, x2s, 64, 0, nullptr, nullptr);
  run_sff(x2s, g48, 48, 48, 1, cat48, sA48, sA48b, sB48, sB48b, at48, fu48);
  k_up2<float><<<eg((size_t)BATCH * 64 * P96), 256, 0, stream>>>(fu48, g96, 64, 48, 48);

  run_sff(x1, g96, 96, 96, 2, cat96, sA96, sA96b, sB96, sB96b, at96, fu96);
  k_up2<float><<<eg((size_t)BATCH * 64 * P192), 256, 0, stream>>>(fu96, out192, 64, 96, 96);

  for (int i = 0; i < FFRAMES; ++i) {
    k_blend_reduce<<<BATCH * 64, 256, 0, stream>>>(out192, mask, i, bsum, 64, 192, 192, 2);
    k_blend_apply<<<eg((size_t)BATCH * 64 * P192), 256, 0, stream>>>(
        out192, mask, i, bsum, mxl, 3, 64, 192, 192, 2);
  }

  k_up2<bf16_t><<<eg((size_t)BATCH * 64 * P384), 256, 0, stream>>>(out192, out384b,
                                                                   64, 192, 192);
  conv(out384b, 64, 384, 384, wbL1, 64, 9, PF(50), 64, conv384, 64, 0, nullptr, nullptr);

  for (int i = 0; i < FFRAMES; ++i) {
    k_blend_reduce<<<BATCH * 64, 256, 0, stream>>>(conv384, mask, i, bsum, 64, 384, 384, 1);
    k_blend_apply<<<eg((size_t)BATCH * 64 * P384), 256, 0, stream>>>(
        conv384, mask, i, bsum, mxl, 4, 64, 384, 384, 1);
  }

  k_relu_cvt<<<eg((size_t)BATCH * 64 * P384), 256, 0, stream>>>(conv384, relu384b,
                                                                BATCH * 64 * P384);
  conv(relu384b, 64, 384, 384, wbL2, 16, 9, PF(52), 1,
       (float*)d_out, 1, 3, nullptr, nullptr);
}
